// MixtureOfExpertsLayer_28321014350374
// MI455X (gfx1250) — matrix operand order repaired and checked by reading, not compiled, not run on hardware
//
#include <hip/hip_runtime.h>

// ---------------- CDNA5 WMMA fragment types ----------------
typedef __attribute__((ext_vector_type(16))) __bf16 v16bf;
typedef __attribute__((ext_vector_type(8)))  __bf16 v8bf;
typedef __attribute__((ext_vector_type(4)))  __bf16 v4bf;
typedef __attribute__((ext_vector_type(2)))  __bf16 v2bf;
typedef __attribute__((ext_vector_type(8)))  float  v8f;

// Pointer types for GLOBAL_LOAD_ASYNC_TO_LDS_B128 (from hipcc diagnostic:
// param is 'vector_size(16) int' pointer; addrspace(1)=global, addrspace(3)=LDS)
typedef int v4i_ __attribute__((vector_size(16)));
typedef __attribute__((address_space(1))) v4i_* g128_t;
typedef __attribute__((address_space(3))) v4i_* l128_t;

// Problem constants (B=2,S=1024 -> T=2048)
constexpr int T  = 2048;   // tokens
constexpr int Hd = 1024;   // hidden
constexpr int Fd = 2048;   // ffn dim
constexpr int E  = 8;      // experts
// top-k = 2 (hardcoded in router)

#if __has_builtin(__builtin_amdgcn_global_load_async_to_lds_b128) && \
    __has_builtin(__builtin_amdgcn_s_wait_asynccnt)
#define HAVE_ASYNC_LDS 1
#else
#define HAVE_ASYNC_LDS 0
#endif

__device__ __forceinline__ __bf16 to_bf(float f) { return (__bf16)f; }

__device__ __forceinline__ float fast_rcp(float x) {
#if __has_builtin(__builtin_amdgcn_rcpf)
  return __builtin_amdgcn_rcpf(x);   // v_rcp_f32, ~1ulp (<< bf16 noise)
#else
  return 1.f / x;
#endif
}

__device__ __forceinline__ v8f zero8() {
  v8f z = {0.f, 0.f, 0.f, 0.f, 0.f, 0.f, 0.f, 0.f};
  return z;
}

// Pack a 4x4 blocklet column (4 consecutive k for one n) as one b64 store.
__device__ __forceinline__ void st4(__bf16* p, float a, float b, float c, float d) {
  v4bf v = { to_bf(a), to_bf(b), to_bf(c), to_bf(d) };
  *(v4bf*)p = v;    // ds_store_b64; pairs lower to v_cvt_pk_bf16_f32
}

// ---------------- 0) init: zero out/count/maps (harness poisons ws/out) ----
__global__ __launch_bounds__(256) void moe_init(float* __restrict__ out,
                                                int* __restrict__ counts,
                                                int* __restrict__ tmap,
                                                float* __restrict__ scal) {
  int i = blockIdx.x * 256 + threadIdx.x;
  if (i < T * Hd) out[i] = 0.f;
  if (i < E * T) { tmap[i] = 0; scal[i] = 0.f; }
  if (i < E) counts[i] = 0;
}

// ---------------- 1) router: logits, top-2, softmax, slot append -----------
// One wave32 per token. 8 dot products of length 1024, wave-reduced.
__global__ __launch_bounds__(256) void moe_router(const float* __restrict__ x,
                                                  const float* __restrict__ rw,
                                                  int* __restrict__ counts,
                                                  int* __restrict__ tmap,
                                                  float* __restrict__ scal) {
  const int lane = threadIdx.x & 31;
  const int t = blockIdx.x * 8 + (threadIdx.x >> 5);

  float acc[E];
#pragma unroll
  for (int e = 0; e < E; ++e) acc[e] = 0.f;

  const float* xr = x + (size_t)t * Hd;
  for (int h = lane; h < Hd; h += 32) {
    float xv = xr[h];
    const float* r = rw + (size_t)h * E;  // router_w is [H,E]
#pragma unroll
    for (int e = 0; e < E; ++e) acc[e] = fmaf(xv, r[e], acc[e]);
  }
#pragma unroll
  for (int e = 0; e < E; ++e) {
#pragma unroll
    for (int off = 16; off > 0; off >>= 1) acc[e] += __shfl_xor(acc[e], off, 32);
  }

  if (lane == 0) {
    // top-2 with first-max-wins (matches jax.lax.top_k tie behavior)
    float b0 = -3.402823466e38f; int i0 = 0;
#pragma unroll
    for (int e = 0; e < E; ++e) if (acc[e] > b0) { b0 = acc[e]; i0 = e; }
    float b1 = -3.402823466e38f; int i1 = 0;
#pragma unroll
    for (int e = 0; e < E; ++e) if (e != i0 && acc[e] > b1) { b1 = acc[e]; i1 = e; }
    // softmax over the two selected logits (b0 >= b1)
    float e1 = __expf(b1 - b0);
    float inv = fast_rcp(1.f + e1);
    float w0 = inv, w1 = e1 * inv;

    int s0 = atomicAdd(&counts[i0], 1);
    tmap[i0 * T + s0] = t; scal[i0 * T + s0] = w0;
    int s1 = atomicAdd(&counts[i1], 1);
    tmap[i1 * T + s1] = t; scal[i1 * T + s1] = w1;
  }
}

// ---------------- 2) grouped GEMM1: act = silu(Xg@W1)*(Xg@W3)*combine ------
// Block: 256 thr (8 waves). Tile: 16 tokens x 256 F cols. K-step 32 (bf16 WMMA).
// Wave w owns cols [w*32, w*32+32): two 16x16 WMMA tiles for W1 and for W3.
__global__ __launch_bounds__(256) void moe_gemm1(const float* __restrict__ x,
                                                 const float* __restrict__ w1,
                                                 const float* __restrict__ w3,
                                                 const int* __restrict__ tmap,
                                                 const float* __restrict__ scal,
                                                 const int* __restrict__ counts,
                                                 __bf16* __restrict__ act) {
  const int e  = blockIdx.z;
  const int mt = blockIdx.y;   // token tile (16 slots)
  const int nt = blockIdx.x;   // F tile (256 cols)
  if (mt * 16 >= counts[e]) return;   // routing-aware early exit (4x FLOP cut)

  __shared__ alignas(64) __bf16 sA [16 * 32];    // A tile  [row][k]
  __shared__ alignas(64) __bf16 sB1[256 * 32];   // W1 tile transposed [n][k]
  __shared__ alignas(64) __bf16 sB3[256 * 32];   // W3 tile transposed [n][k]
  __shared__ int   sTok[16];
  __shared__ float sScl[16];

  const int tid  = threadIdx.x;
  const int lane = tid & 31;
  const int wave = tid >> 5;
  const int half = lane >> 4;    // A: K-half selector / B: K-half / C: M-half
  const int l15  = lane & 15;

  if (tid < 16) {
    int s = e * T + mt * 16 + tid;
    sTok[tid] = tmap[s];
    sScl[tid] = scal[s];
  }
  __syncthreads();

  const int nb = nt * 256;
  const float* W1 = w1 + (size_t)e * Hd * Fd + nb;  // [H,F] panel, col offset nb
  const float* W3 = w3 + (size_t)e * Hd * Fd + nb;

  // A staging ownership: row = tid>>4, two consecutive k = (tid&15)*2.
  // Hoist the gathered row pointer out of the K-loop (no per-iter ds_load/addr math).
  const int arow = tid >> 4;
  const int ak   = (tid & 15) * 2;
  const float* xrow = x + (size_t)sTok[arow] * Hd;

  v8f c1a = zero8(), c1b = zero8(), c3a = zero8(), c3b = zero8();

  for (int k0 = 0; k0 < Hd; k0 += 32) {
    // ---- stage A: float2 load, pack, one ds_store_b32 per thread
    {
      float2 xf = *(const float2*)(xrow + k0 + ak);
      v2bf p = { to_bf(xf.x), to_bf(xf.y) };
      *(v2bf*)&sA[arow * 32 + ak] = p;
    }
    // ---- stage B: 4x4 blocklets. 4 float4 loads from 4 consecutive k-rows
    // (128b coalesced), register transpose, 4 packed b64 LDS stores.
    if (k0 + 32 < Hd) {
      __builtin_prefetch(&W1[(size_t)(k0 + 32) * Fd + tid], 0, 1);
      __builtin_prefetch(&W3[(size_t)(k0 + 32) * Fd + tid], 0, 1);
    }
#pragma unroll
    for (int r = 0; r < 2; ++r) {
      int bid = tid + r * 256;        // blocklet id: 8 k-groups x 64 n4-groups
      int kb  = (bid >> 6) * 4;       // k base (0..28)
      int n4  = bid & 63;             // float4 column group, n = n4*4
      const float* b1p = W1 + (size_t)(k0 + kb) * Fd;
      float4 a0 = ((const float4*)(b1p         ))[n4];
      float4 a1 = ((const float4*)(b1p + Fd    ))[n4];
      float4 a2 = ((const float4*)(b1p + 2 * Fd))[n4];
      float4 a3 = ((const float4*)(b1p + 3 * Fd))[n4];
      __bf16* d1 = &sB1[(n4 * 4) * 32 + kb];
      st4(d1 + 0 * 32, a0.x, a1.x, a2.x, a3.x);
      st4(d1 + 1 * 32, a0.y, a1.y, a2.y, a3.y);
      st4(d1 + 2 * 32, a0.z, a1.z, a2.z, a3.z);
      st4(d1 + 3 * 32, a0.w, a1.w, a2.w, a3.w);
      const float* b3p = W3 + (size_t)(k0 + kb) * Fd;
      a0 = ((const float4*)(b3p         ))[n4];
      a1 = ((const float4*)(b3p + Fd    ))[n4];
      a2 = ((const float4*)(b3p + 2 * Fd))[n4];
      a3 = ((const float4*)(b3p + 3 * Fd))[n4];
      __bf16* d3 = &sB3[(n4 * 4) * 32 + kb];
      st4(d3 + 0 * 32, a0.x, a1.x, a2.x, a3.x);
      st4(d3 + 1 * 32, a0.y, a1.y, a2.y, a3.y);
      st4(d3 + 2 * 32, a0.z, a1.z, a2.z, a3.z);
      st4(d3 + 3 * 32, a0.w, a1.w, a2.w, a3.w);
    }
    __syncthreads();

    // ---- A fragment, 16x32: lane(0-15)=row, elems e -> k=16*half+e
    // (same k order as B) => one contiguous 32B LDS chunk.
    v16bf a = *(const v16bf*)&sA[l15 * 32 + half * 16];
    // ---- B fragment, 32x16: lane n=l15, khalf=half, elems e -> k=16*half+e
    // => one contiguous 32B chunk of transposed LDS tile.
    const int n0 = wave * 32 + l15;
    v16bf b;
    b = *(const v16bf*)&sB1[(n0     ) * 32 + half * 16];
    c1a = __builtin_amdgcn_wmma_f32_16x16x32_bf16(false, a, false, b, (short)0, c1a, false, false);
    b = *(const v16bf*)&sB1[(n0 + 16) * 32 + half * 16];
    c1b = __builtin_amdgcn_wmma_f32_16x16x32_bf16(false, a, false, b, (short)0, c1b, false, false);
    b = *(const v16bf*)&sB3[(n0     ) * 32 + half * 16];
    c3a = __builtin_amdgcn_wmma_f32_16x16x32_bf16(false, a, false, b, (short)0, c3a, false, false);
    b = *(const v16bf*)&sB3[(n0 + 16) * 32 + half * 16];
    c3b = __builtin_amdgcn_wmma_f32_16x16x32_bf16(false, a, false, b, (short)0, c3b, false, false);
    __syncthreads();
  }

  // ---- epilogue: silu(gate)*up * combine_weight, store bf16 act
  // C layout: VGPR r -> row = r + 8*half, col = l15 (within 16-wide subtile)
#pragma unroll
  for (int s = 0; s < 2; ++s) {
    v8f cg = s ? c1b : c1a;
    v8f cu = s ? c3b : c3a;
    int col = nb + wave * 32 + s * 16 + l15;
#pragma unroll
    for (int r = 0; r < 8; ++r) {
      int m = r + 8 * half;
      float g = cg[r];
      float sig = fast_rcp(1.f + __expf(-g));          // v_rcp_f32, no IEEE div chain
      float v = g * sig * cu[r] * sScl[m];             // padding rows: scale=0
      size_t row = (size_t)(e * T + mt * 16 + m);
      act[row * Fd + col] = to_bf(v);
    }
  }
}

// ---------------- 3) grouped GEMM2: out[token] += act @ W2[e] --------------
// Tile: 16 slots x 256 H cols, K over F=2048 in steps of 32.
// A tile (already bf16) is fetched with GLOBAL_LOAD_ASYNC_TO_LDS_B128
// (ASYNCcnt-tracked DMA into LDS) when the builtin exists.
__global__ __launch_bounds__(256) void moe_gemm2(const __bf16* __restrict__ act,
                                                 const float* __restrict__ w2,
                                                 const int* __restrict__ tmap,
                                                 const int* __restrict__ counts,
                                                 float* __restrict__ out) {
  const int e  = blockIdx.z;
  const int mt = blockIdx.y;
  const int nt = blockIdx.x;   // H tile (256 cols)
  if (mt * 16 >= counts[e]) return;

  __shared__ alignas(64) __bf16 sA[16 * 32];
  __shared__ alignas(64) __bf16 sB[256 * 32];   // W2 tile transposed [n][k]
  __shared__ int sTok[16];

  const int tid  = threadIdx.x;
  const int lane = tid & 31;
  const int wave = tid >> 5;
  const int half = lane >> 4;
  const int l15  = lane & 15;

  if (tid < 16) sTok[tid] = tmap[e * T + mt * 16 + tid];
  __syncthreads();

  const int nb = nt * 256;
  const float* W2 = w2 + (size_t)e * Fd * Hd + nb;               // [F,H] panel
  const __bf16* Ar = act + (size_t)(e * T + mt * 16) * Fd;       // 16 act rows

  v8f ca = zero8(), cb = zero8();

  for (int k0 = 0; k0 < Fd; k0 += 32) {
    // ---- stage A: 16 rows x 32 k bf16 (1KB) -> LDS
#if HAVE_ASYNC_LDS
    if (tid < 32) {
      // 64 chunks of 16B; lane handles chunk tid and tid+32.
      // chunk c: row = c>>2, k-offset = (c&3)*8, LDS elem offset = 8c.
      int c0 = tid, c1 = tid + 32;
      const __bf16* g0 = Ar + (size_t)(c0 >> 2) * Fd + k0 + (c0 & 3) * 8;
      const __bf16* g1 = Ar + (size_t)(c1 >> 2) * Fd + k0 + (c1 & 3) * 8;
      __builtin_amdgcn_global_load_async_to_lds_b128(
          (g128_t)(uintptr_t)g0, (l128_t)(uintptr_t)&sA[c0 * 8], 0, 0);
      __builtin_amdgcn_global_load_async_to_lds_b128(
          (g128_t)(uintptr_t)g1, (l128_t)(uintptr_t)&sA[c1 * 8], 0, 0);
    }
#else
    {
      int row = tid >> 4, k = (tid & 15) * 2;
      *(unsigned int*)&sA[row * 32 + k] =
          *(const unsigned int*)&Ar[(size_t)row * Fd + k0 + k];
    }
#endif
    // ---- stage B: 4x4 blocklets (see GEMM1)
    if (k0 + 32 < Fd)
      __builtin_prefetch(&W2[(size_t)(k0 + 32) * Hd + tid], 0, 1);
#pragma unroll
    for (int r = 0; r < 2; ++r) {
      int bid = tid + r * 256;
      int kb  = (bid >> 6) * 4;
      int n4  = bid & 63;
      const float* bp = W2 + (size_t)(k0 + kb) * Hd;
      float4 a0 = ((const float4*)(bp         ))[n4];
      float4 a1 = ((const float4*)(bp + Hd    ))[n4];
      float4 a2 = ((const float4*)(bp + 2 * Hd))[n4];
      float4 a3 = ((const float4*)(bp + 3 * Hd))[n4];
      __bf16* d = &sB[(n4 * 4) * 32 + kb];
      st4(d + 0 * 32, a0.x, a1.x, a2.x, a3.x);
      st4(d + 1 * 32, a0.y, a1.y, a2.y, a3.y);
      st4(d + 2 * 32, a0.z, a1.z, a2.z, a3.z);
      st4(d + 3 * 32, a0.w, a1.w, a2.w, a3.w);
    }
#if HAVE_ASYNC_LDS
    __builtin_amdgcn_s_wait_asynccnt(0);   // async A-tile landed in LDS
#endif
    __syncthreads();

    v16bf a = *(const v16bf*)&sA[l15 * 32 + half * 16];
    const int n0 = wave * 32 + l15;
    v16bf b;
    b = *(const v16bf*)&sB[(n0     ) * 32 + half * 16];
    ca = __builtin_amdgcn_wmma_f32_16x16x32_bf16(false, a, false, b, (short)0, ca, false, false);
    b = *(const v16bf*)&sB[(n0 + 16) * 32 + half * 16];
    cb = __builtin_amdgcn_wmma_f32_16x16x32_bf16(false, a, false, b, (short)0, cb, false, false);
    __syncthreads();
  }

  // scatter-accumulate to out[token]. Exactly 2 adds per (t,h) across all
  // expert blocks and f32 addition is commutative -> deterministic result.
#pragma unroll
  for (int s = 0; s < 2; ++s) {
    v8f c = s ? cb : ca;
    int col = nb + wave * 32 + s * 16 + l15;
#pragma unroll
    for (int r = 0; r < 8; ++r) {
      int m = r + 8 * half;
      int tok = sTok[m];                               // padding rows add 0 to out[0]
      unsafeAtomicAdd(&out[(size_t)tok * Hd + col], c[r]);
    }
  }
}

// ---------------- host-side launch -----------------------------------------
extern "C" void kernel_launch(void* const* d_in, const int* in_sizes, int n_in,
                              void* d_out, int out_size, void* d_ws, size_t ws_size,
                              hipStream_t stream) {
  const float* x  = (const float*)d_in[0];   // [B,S,H] == [T,H]
  const float* rw = (const float*)d_in[1];   // [H,E]
  const float* w1 = (const float*)d_in[2];   // [E,H,F]
  const float* w3 = (const float*)d_in[3];   // [E,H,F]
  const float* w2 = (const float*)d_in[4];   // [E,F,H]
  float* out = (float*)d_out;                // [T,H] f32

  // workspace layout (~64.2 MB): counts | token_map | scales | act(bf16)
  char* ws = (char*)d_ws;
  int*    counts = (int*)ws;                                   // 8 ints
  int*    tmap   = (int*)(ws + 256);                           // E*T ints (64KB)
  float*  scal   = (float*)(ws + 256 + (size_t)E * T * 4);     // E*T f32  (64KB)
  __bf16* act    = (__bf16*)(ws + 256 + (size_t)E * T * 8);    // E*T*F bf16 (64MB)

  moe_init<<<(T * Hd) / 256, 256, 0, stream>>>(out, counts, tmap, scal);
  moe_router<<<T / 8, 256, 0, stream>>>(x, rw, counts, tmap, scal);
  moe_gemm1<<<dim3(Fd / 256, T / 16, E), 256, 0, stream>>>(x, w1, w3, tmap, scal, counts, act);
  moe_gemm2<<<dim3(Hd / 256, T / 16, E), 256, 0, stream>>>(act, w2, tmap, counts, out);
}
